// LSTM_7121055776971
// MI455X (gfx1250) — compile-verified, matrix operand order checked by reading, not run on hardware
//
#include <hip/hip_runtime.h>
#include <hip/hip_bf16.h>

// ---------------- dims ----------------
#define IN_DIM 512
#define H_DIM  1024
#define T_LEN  256
#define BS     128
#define ZIN    (IN_DIM + H_DIM)      // 1536
#define G4     (4 * H_DIM)           // 4096
#define KCHUNK 64
#define NSTAGE (ZIN / KCHUNK)        // 24

typedef __attribute__((ext_vector_type(16))) __bf16 v16bf;
typedef __attribute__((ext_vector_type(8)))  float  v8f;

// ---------------- prep kernels ----------------

// Stack Wf/Wi/Wc/Wo -> bf16 [4096][1536], gate-major (f,i,c,o)
__global__ void prep_w(const float* __restrict__ Wf, const float* __restrict__ Wi,
                       const float* __restrict__ Wc, const float* __restrict__ Wo,
                       __hip_bfloat16* __restrict__ out) {
    int idx = blockIdx.x * blockDim.x + threadIdx.x;
    if (idx >= G4 * ZIN) return;
    int j = idx / ZIN;
    int k = idx - j * ZIN;
    int gate = j >> 10;          // /1024
    int r    = j & 1023;
    const float* W = (gate == 0) ? Wf : (gate == 1) ? Wi : (gate == 2) ? Wc : Wo;
    out[idx] = __float2bfloat16(W[r * ZIN + k]);
}

// x fp32 -> bf16, flat
__global__ void prep_x(const float* __restrict__ x, __hip_bfloat16* __restrict__ xb, int n) {
    int idx = blockIdx.x * blockDim.x + threadIdx.x;
    if (idx < n) xb[idx] = __float2bfloat16(x[idx]);
}

// h -> bf16 state, c -> fp32 state copy (never mutate inputs)
__global__ void prep_state(const float* __restrict__ h, const float* __restrict__ c,
                           __hip_bfloat16* __restrict__ hb, float* __restrict__ cs) {
    int idx = blockIdx.x * blockDim.x + threadIdx.x;
    if (idx >= BS * H_DIM) return;
    hb[idx] = __float2bfloat16(h[idx]);
    cs[idx] = c[idx];
}

// ---------------- WMMA gate GEMM ----------------
// g[128][4096] = [h|x] @ W.T   (bias added in gate kernel)
// grid: 64 blocks (N-tiles of 64), block: 256 threads = 8 waves.
// wave w computes rows 16w..16w+15, cols n0..n0+63 (4 accumulators).
// B tiles (64 n x 64 k bf16 = 8 KB) are double-buffered in LDS via
// global_load_async_to_lds_b128 (ASYNCcnt), shared by all 8 waves.

__device__ __forceinline__ v16bf load_frag(const __hip_bfloat16* p, int hi) {
    // 16-bit A/B fragment: elements 0..7 = K in [hi*8, hi*8+7],
    // elements 8..15 = K in [16+hi*8, 16+hi*8+7]  (two 16B loads)
    union { v16bf v; uint4 u[2]; } f;
    const char* cp = (const char*)p + hi * 16;
    f.u[0] = *(const uint4*)(cp);
    f.u[1] = *(const uint4*)(cp + 32);
    return f.v;
}

__global__ void __launch_bounds__(256)
gemm_step(const __hip_bfloat16* __restrict__ Wb,  // [4096][1536]
          const __hip_bfloat16* __restrict__ hb,  // [128][1024]
          const __hip_bfloat16* __restrict__ xb,  // [128][512] (step t)
          float* __restrict__ g) {                // [128][4096]
    __shared__ __hip_bfloat16 smB[2][64 * KCHUNK];   // 2 x 8 KB, [n][k] row-major

    const int tid  = threadIdx.x;
    const int lane = tid & 31;
    const int wave = tid >> 5;
    const int n0   = blockIdx.x * 64;
    const int m0   = wave * 16;
    const int mrow = m0 + (lane & 15);
    const int hi   = lane >> 4;
    const int ncol = lane & 15;

    // low 32 bits of a generic pointer to a __shared__ object == LDS byte offset
    const uint32_t ldsBase = (uint32_t)(uintptr_t)(&smB[0][0]);

    // Issue one stage's B tile: 512 x 16B segments, 2 async copies per thread.
    auto issue_stage = [&](int k0, int buf) {
#pragma unroll
        for (int i = 0; i < 2; ++i) {
            int e   = tid + i * 256;   // 0..511
            int row = e >> 3;          // n within tile: 0..63
            int seg = e & 7;           // 16B segment within row: 0..7
            const __hip_bfloat16* gp = Wb + (size_t)(n0 + row) * ZIN + k0 + seg * 8;
            uint32_t ldst = ldsBase + (uint32_t)buf * (64 * KCHUNK * 2)
                          + (uint32_t)(row * (KCHUNK * 2) + seg * 16);
            asm volatile("global_load_async_to_lds_b128 %0, %1, off"
                         :: "v"(ldst), "v"(gp)
                         : "memory");
        }
    };

    v8f acc[4] = {};

    issue_stage(0, 0);

    for (int s = 0; s < NSTAGE; ++s) {
        const int k0  = s * KCHUNK;
        const int buf = s & 1;

        if (s + 1 < NSTAGE) {
            issue_stage((s + 1) * KCHUNK, buf ^ 1);
            // stage s (2 asyncs) retired; stage s+1 (2 asyncs) may be in flight
            asm volatile("s_wait_asynccnt 2" ::: "memory");
        } else {
            asm volatile("s_wait_asynccnt 0" ::: "memory");
        }
        __syncthreads();   // B tile for stage s visible to all waves

        // A source: recurrent h for K<1024, input x_t for K>=1024
        const __hip_bfloat16* ap = (k0 < H_DIM)
            ? hb + (size_t)mrow * H_DIM + k0
            : xb + (size_t)mrow * IN_DIM + (k0 - H_DIM);

        const __hip_bfloat16* bbase = &smB[buf][0];
#pragma unroll
        for (int ks = 0; ks < KCHUNK; ks += 32) {
            v16bf a = load_frag(ap + ks, hi);
#pragma unroll
            for (int gi = 0; gi < 4; ++gi) {
                const __hip_bfloat16* bp = bbase + (gi * 16 + ncol) * KCHUNK + ks;
                v16bf b = load_frag(bp, hi);
                acc[gi] = __builtin_amdgcn_wmma_f32_16x16x32_bf16(
                    false, a, false, b, (short)0, acc[gi], false, false);
            }
        }
        __syncthreads();   // everyone done reading buf before it is refilled
    }

    // D layout: VGPR r -> M = m0 + r + 8*hi, N = n0 + gi*16 + ncol
    const int mbase = m0 + hi * 8;
#pragma unroll
    for (int gi = 0; gi < 4; ++gi) {
        const int n = n0 + gi * 16 + ncol;
#pragma unroll
        for (int r = 0; r < 8; ++r) {
            g[(size_t)(mbase + r) * G4 + n] = acc[gi][r];
        }
    }
}

// ---------------- gate / softmax / state update ----------------
// one block (256 threads, 8 waves) per batch row

__device__ __forceinline__ float blk_max(float v, float* sm) {
#pragma unroll
    for (int o = 16; o > 0; o >>= 1) v = fmaxf(v, __shfl_xor(v, o, 32));
    if ((threadIdx.x & 31) == 0) sm[threadIdx.x >> 5] = v;
    __syncthreads();
    float r = sm[0];
#pragma unroll
    for (int i = 1; i < 8; ++i) r = fmaxf(r, sm[i]);
    __syncthreads();
    return r;
}

__device__ __forceinline__ float blk_sum(float v, float* sm) {
#pragma unroll
    for (int o = 16; o > 0; o >>= 1) v += __shfl_xor(v, o, 32);
    if ((threadIdx.x & 31) == 0) sm[threadIdx.x >> 5] = v;
    __syncthreads();
    float r = 0.f;
#pragma unroll
    for (int i = 1; i < 8; ++i) r += sm[i];
    r += sm[0];
    __syncthreads();
    return r;
}

__device__ __forceinline__ void softmax4(const float* __restrict__ base,
                                         const float* __restrict__ bias,
                                         float* __restrict__ outv, float* sm) {
    float v[4];
    float mx = -3.0e38f;
#pragma unroll
    for (int k = 0; k < 4; ++k) {
        int j = threadIdx.x + k * 256;
        v[k] = base[j] + bias[j];
        mx = fmaxf(mx, v[k]);
    }
    mx = blk_max(mx, sm);
    float s = 0.f;
#pragma unroll
    for (int k = 0; k < 4; ++k) { v[k] = expf(v[k] - mx); s += v[k]; }
    s = blk_sum(s, sm);
    float inv = 1.0f / s;
#pragma unroll
    for (int k = 0; k < 4; ++k) outv[k] = v[k] * inv;
}

__global__ void gate_step(const float* __restrict__ g,            // [128][4096]
                          const float* __restrict__ bf_, const float* __restrict__ bi_,
                          const float* __restrict__ bc_, const float* __restrict__ bo_,
                          float* __restrict__ cs,                 // [128][1024] fp32 state
                          __hip_bfloat16* __restrict__ hb,        // [128][1024] bf16 state
                          float* __restrict__ out_t) {            // out[t]: [128][1024]
    __shared__ float sm[8];
    const int b = blockIdx.x;
    const float* gr = g + (size_t)b * G4;

    float fv[4], iv[4], ov[4];
    softmax4(gr + 0 * H_DIM, bf_, fv, sm);   // forget gate
    softmax4(gr + 1 * H_DIM, bi_, iv, sm);   // input gate
    softmax4(gr + 3 * H_DIM, bo_, ov, sm);   // output gate

#pragma unroll
    for (int k = 0; k < 4; ++k) {
        int j = threadIdx.x + k * 256;
        float ct = tanhf(gr[2 * H_DIM + j] + bc_[j]);
        size_t sidx = (size_t)b * H_DIM + j;
        float cn = fv[k] * cs[sidx] + iv[k] * ct;
        cs[sidx] = cn;
        float hn = ov[k] * tanhf(cn);
        out_t[sidx] = hn;
        hb[sidx] = __float2bfloat16(hn);
    }
}

// ---------------- host launcher ----------------

extern "C" void kernel_launch(void* const* d_in, const int* in_sizes, int n_in,
                              void* d_out, int out_size, void* d_ws, size_t ws_size,
                              hipStream_t stream) {
    const float* x  = (const float*)d_in[0];   // [256,128,512]
    const float* h  = (const float*)d_in[1];   // [128,1024]
    const float* c  = (const float*)d_in[2];   // [128,1024]
    const float* Wf = (const float*)d_in[3];
    const float* bf = (const float*)d_in[4];
    const float* Wi = (const float*)d_in[5];
    const float* bi = (const float*)d_in[6];
    const float* Wc = (const float*)d_in[7];
    const float* bc = (const float*)d_in[8];
    const float* Wo = (const float*)d_in[9];
    const float* bo = (const float*)d_in[10];
    float* out = (float*)d_out;

    char* ws = (char*)d_ws;
    // workspace layout (all offsets 256B aligned)
    __hip_bfloat16* Wb = (__hip_bfloat16*)(ws);                       // 4096*1536*2 = 12,582,912
    __hip_bfloat16* xb = (__hip_bfloat16*)(ws + 12582912);            // 256*128*512*2 = 33,554,432
    __hip_bfloat16* hb = (__hip_bfloat16*)(ws + 12582912 + 33554432); // 128*1024*2 = 262,144
    float* cs   = (float*)(ws + 46399488);                            // 128*1024*4 = 524,288
    float* gbuf = (float*)(ws + 46923776);                            // 128*4096*4 = 2,097,152

    // one-time prep
    prep_w<<<(G4 * ZIN + 255) / 256, 256, 0, stream>>>(Wf, Wi, Wc, Wo, Wb);
    int nx = T_LEN * BS * IN_DIM;
    prep_x<<<(nx + 255) / 256, 256, 0, stream>>>(x, xb, nx);
    prep_state<<<(BS * H_DIM + 255) / 256, 256, 0, stream>>>(h, c, hb, cs);

    // sequential recurrence
    for (int t = 0; t < T_LEN; ++t) {
        gemm_step<<<G4 / 64, 256, 0, stream>>>(Wb, hb, xb + (size_t)t * BS * IN_DIM, gbuf);
        gate_step<<<BS, 256, 0, stream>>>(gbuf, bf, bi, bc, bo, cs, hb,
                                          out + (size_t)t * BS * H_DIM);
    }
}
